// NextBlockLSTM_62852551409783
// MI455X (gfx1250) — compile-verified, matrix operand order checked by reading, not run on hardware
//
#include <hip/hip_runtime.h>
#include <hip/hip_bf16.h>
#include <math.h>

// ---------------------------------------------------------------------------
// BiLSTM: T=4096, D=1024, H=1024, x_dim=2048, gates=4H=4096
// Phase 1: Z = Vfor @ w_x  (bf16 WMMA, fp32 accum).  zx[1,t] == Z[T-1-t].
// Phase 2: persistent scan kernel, grid barrier per step, w_h L2-resident.
// ---------------------------------------------------------------------------

#define T_STEPS 4096
#define DIM     1024
#define XDIM    2048
#define GDIM    4096   // 4*H
#define NWG_SCAN 128   // 2 dirs * 64 blocks

typedef __attribute__((ext_vector_type(16))) __bf16 v16bf;
typedef __attribute__((ext_vector_type(8)))  __bf16 v8bf;
typedef __attribute__((ext_vector_type(8)))  float  v8f;

// ------------------------- helpers -----------------------------------------

__device__ __forceinline__ unsigned short f2bf(float f) {
    unsigned int x = __float_as_uint(f);
    // round-to-nearest-even
    unsigned int r = x + 0x7FFFu + ((x >> 16) & 1u);
    return (unsigned short)(r >> 16);
}

// 16-bit A/B fragment loader (CDNA5 WMMA 16x16x32 layout):
// lane&15 = row (M for A / N for Bt), half = lane>>4.
// Lane holds K = {half*8 .. half*8+7} U {16+half*8 .. 16+half*8+7}
// as two contiguous 8-element (16-byte) chunks.
__device__ __forceinline__ v16bf load_frag(const unsigned short* __restrict__ base,
                                           int row, int kk, int half) {
    const unsigned short* p = base + (size_t)row * XDIM + kk + half * 8;
    v8bf lo = *(const v8bf*)(p);
    v8bf hi = *(const v8bf*)(p + 16);
    v16bf f;
#pragma unroll
    for (int i = 0; i < 8; ++i) { f[i] = lo[i]; f[8 + i] = hi[i]; }
    return f;
}

// ------------------------- init --------------------------------------------

__global__ void bilstm_init(float* __restrict__ hbuf, int* __restrict__ bar) {
    int i = blockIdx.x * blockDim.x + threadIdx.x;
    if (i < 4 * DIM) hbuf[i] = 0.0f;   // both parities, both directions
    if (i < 2) bar[i] = 0;
}

// ------------------------- packing -----------------------------------------

// Vfor[t][j] = j<1024 ? res_for[t][j] : res_back[T-1-t][j-1024]   (bf16)
__global__ void bilstm_pack_vfor(const float* __restrict__ res_for,
                                 const float* __restrict__ res_back,
                                 unsigned short* __restrict__ vfor) {
    int idx = blockIdx.x * blockDim.x + threadIdx.x;   // T*XDIM threads
    int t = idx >> 11;          // /2048
    int j = idx & 2047;
    float v = (j < DIM) ? res_for[(size_t)t * DIM + j]
                        : res_back[(size_t)(T_STEPS - 1 - t) * DIM + (j - DIM)];
    vfor[idx] = f2bf(v);
}

// wxT[n][k] = w_x[k][n] = w[k*4096 + n], k<2048   (bf16, transposed)
__global__ void bilstm_pack_wxT(const float* __restrict__ w,
                                unsigned short* __restrict__ wxT) {
    int idx = blockIdx.x * blockDim.x + threadIdx.x;   // GDIM*XDIM threads
    int n = idx >> 11;          // /2048
    int k = idx & 2047;
    wxT[(size_t)n * XDIM + k] = f2bf(w[(size_t)k * GDIM + n]);
}

// ------------------------- phase 1: WMMA GEMM -------------------------------
// Z(4096x4096) = Vfor(4096x2048) @ w_x(2048x4096); each wave: 16x64 strip.
// Fragment loads are software-pipelined one K-step ahead so v_wmma never
// drains LOADcnt to zero (next-iteration global_load_b128s stay in flight
// behind the matrix ops).
__global__ void bilstm_gemm_zx(const unsigned short* __restrict__ A,   // Vfor
                               const unsigned short* __restrict__ Bt,  // wxT
                               float* __restrict__ Z) {
    int wave  = blockIdx.x * (blockDim.x >> 5) + (threadIdx.x >> 5);
    int lane  = threadIdx.x & 31;
    int mtile = wave >> 6;      // 0..255 (M/16)
    int ngrp  = wave & 63;      // 0..63  (N/64)
    int half  = lane >> 4;
    int rrow  = lane & 15;

    const int arow = mtile * 16 + rrow;
    const int brow = ngrp * 64 + rrow;

    v8f acc0 = {}, acc1 = {}, acc2 = {}, acc3 = {};

    // prologue: fragments for kk = 0
    v16bf a  = load_frag(A,  arow,      0, half);
    v16bf b0 = load_frag(Bt, brow +  0, 0, half);
    v16bf b1 = load_frag(Bt, brow + 16, 0, half);
    v16bf b2 = load_frag(Bt, brow + 32, 0, half);
    v16bf b3 = load_frag(Bt, brow + 48, 0, half);

    for (int kk = 0; kk < XDIM; kk += 32) {
        // prefetch next K-step (wraps to 0 on last iter; harmless re-load)
        const int kn = (kk + 32 < XDIM) ? (kk + 32) : 0;
        v16bf an  = load_frag(A,  arow,      kn, half);
        v16bf bn0 = load_frag(Bt, brow +  0, kn, half);
        v16bf bn1 = load_frag(Bt, brow + 16, kn, half);
        v16bf bn2 = load_frag(Bt, brow + 32, kn, half);
        v16bf bn3 = load_frag(Bt, brow + 48, kn, half);

        acc0 = __builtin_amdgcn_wmma_f32_16x16x32_bf16(false, a, false, b0,
                                                       (short)0, acc0, false, false);
        acc1 = __builtin_amdgcn_wmma_f32_16x16x32_bf16(false, a, false, b1,
                                                       (short)0, acc1, false, false);
        acc2 = __builtin_amdgcn_wmma_f32_16x16x32_bf16(false, a, false, b2,
                                                       (short)0, acc2, false, false);
        acc3 = __builtin_amdgcn_wmma_f32_16x16x32_bf16(false, a, false, b3,
                                                       (short)0, acc3, false, false);

        a = an; b0 = bn0; b1 = bn1; b2 = bn2; b3 = bn3;
    }

    // C/D layout: VGPR r -> M = half*8 + r ; lane&15 -> N
#pragma unroll
    for (int r = 0; r < 8; ++r) {
        int M = mtile * 16 + half * 8 + r;
        size_t rowoff = (size_t)M * GDIM + (size_t)ngrp * 64 + rrow;
        Z[rowoff +  0] = acc0[r];
        Z[rowoff + 16] = acc1[r];
        Z[rowoff + 32] = acc2[r];
        Z[rowoff + 48] = acc3[r];
    }
}

// ------------------------- phase 2: persistent scan --------------------------
// grid: 128 WGs = 2 dirs x 64 blocks, 256 threads each.
// WG (d,b): hidden units u in [b*16, b*16+16) -> 64 gate columns g*1024+b*16+u.

__device__ __forceinline__ float sigmoidf(float x) {
    return 1.0f / (1.0f + __expf(-x));
}

__global__ void bilstm_scan(const float* __restrict__ w,     // full (3072,4096)
                            const float* __restrict__ Z,     // (4096,4096)
                            float* __restrict__ hbuf,        // [2 parity][2 dir][1024]
                            int* __restrict__ bar,
                            float* __restrict__ out) {       // [2][T][1024]
    const float* __restrict__ wh = w + (size_t)XDIM * GDIM;  // w_h (1024,4096)

    const int d   = blockIdx.x >> 6;   // direction
    const int b   = blockIdx.x & 63;   // hidden-unit block
    const int tid = threadIdx.x;
    const int q   = tid >> 6;          // K quarter 0..3
    const int c   = tid & 63;          // column within WG
    const int g   = c >> 4;            // gate 0..3
    const int u   = c & 15;            // unit within block
    const int col = g * DIM + b * 16 + u;

    __shared__ float hsh[DIM];
    __shared__ float red[4][64];
    __shared__ float zsh[64];
    __shared__ float cst[16];
    if (tid < 16) cst[tid] = 0.0f;

    for (int t = 0; t < T_STEPS; ++t) {
        const int rp = t & 1, wp = rp ^ 1;

        // broadcast h_{t-1} for this direction into LDS
        const float* __restrict__ hprev = hbuf + (size_t)(rp * 2 + d) * DIM;
#pragma unroll
        for (int r = 0; r < 4; ++r) hsh[tid + 256 * r] = hprev[tid + 256 * r];
        __syncthreads();

        const int zrow = d ? (T_STEPS - 1 - t) : t;
        float p = (q == 0) ? Z[(size_t)zrow * GDIM + col] : 0.0f;

        const float* __restrict__ wcol = wh + col;
        const int k0 = q * 256;
#pragma unroll 8
        for (int k = 0; k < 256; ++k)
            p = fmaf(hsh[k0 + k], wcol[(size_t)(k0 + k) * GDIM], p);

        red[q][c] = p;
        __syncthreads();
        if (tid < 64)
            zsh[tid] = red[0][tid] + red[1][tid] + red[2][tid] + red[3][tid];
        __syncthreads();

        if (tid < 16) {
            float zi = zsh[tid];
            float zc = zsh[16 + tid];
            float zf = zsh[32 + tid];
            float zo = zsh[48 + tid];
            float i  = sigmoidf(zi);
            float ci = tanhf(zc);
            float f  = sigmoidf(zf + 1.0f);            // FORGET_BIAS
            float cs = fmaf(f, cst[tid], i * ci);
            cs = fminf(3.0f, fmaxf(-3.0f, cs));        // CELL_CLIP
            float o  = sigmoidf(zo);
            float h  = o * tanhf(cs);
            cst[tid] = cs;
            hbuf[(size_t)(wp * 2 + d) * DIM + b * 16 + tid] = h;
            out[(size_t)d * T_STEPS * DIM + (size_t)t * DIM + b * 16 + tid] = h;
        }

        // device-wide sense-reversing barrier (one per timestep)
        __syncthreads();
        if (tid == 0) {
            __threadfence();
            int gen  = __hip_atomic_load(&bar[1], __ATOMIC_RELAXED,
                                         __HIP_MEMORY_SCOPE_AGENT);
            int prev = __hip_atomic_fetch_add(&bar[0], 1, __ATOMIC_ACQ_REL,
                                              __HIP_MEMORY_SCOPE_AGENT);
            if (prev == NWG_SCAN - 1) {
                __hip_atomic_store(&bar[0], 0, __ATOMIC_RELAXED,
                                   __HIP_MEMORY_SCOPE_AGENT);
                __hip_atomic_store(&bar[1], gen + 1, __ATOMIC_RELEASE,
                                   __HIP_MEMORY_SCOPE_AGENT);
            } else {
                while (__hip_atomic_load(&bar[1], __ATOMIC_ACQUIRE,
                                         __HIP_MEMORY_SCOPE_AGENT) == gen)
                    __builtin_amdgcn_s_sleep(8);
            }
        }
        __syncthreads();
    }
}

// ------------------------- launcher -----------------------------------------

extern "C" void kernel_launch(void* const* d_in, const int* in_sizes, int n_in,
                              void* d_out, int out_size, void* d_ws, size_t ws_size,
                              hipStream_t stream) {
    const float* res_for  = (const float*)d_in[0];   // (1, 4096, 1024) f32
    const float* res_back = (const float*)d_in[1];   // (1, 4096, 1024) f32
    const float* w        = (const float*)d_in[2];   // (3072, 4096)   f32
    float* out            = (float*)d_out;           // 2 * 4096 * 1024 f32

    char* ws = (char*)d_ws;
    unsigned short* vfor = (unsigned short*)(ws);                       // 16 MB
    unsigned short* wxT  = (unsigned short*)(ws + ((size_t)16 << 20));  // 16 MB
    float*          Zbuf = (float*)(ws + ((size_t)32 << 20));           // 64 MB
    float*          hbuf = (float*)(ws + ((size_t)96 << 20));           // 16 KB
    int*            bar  = (int*)(ws + ((size_t)96 << 20) + 4 * DIM * sizeof(float));

    // re-init persistent state every call (deterministic graph replay)
    bilstm_init<<<16, 256, 0, stream>>>(hbuf, bar);

    // pack inputs to bf16 (Vfor) and transposed bf16 w_x
    bilstm_pack_vfor<<<(T_STEPS * XDIM) / 256, 256, 0, stream>>>(res_for, res_back, vfor);
    bilstm_pack_wxT<<<(GDIM * XDIM) / 256, 256, 0, stream>>>(w, wxT);

    // Phase 1: Z = Vfor @ w_x via v_wmma_f32_16x16x32_bf16
    // jobs = 256 M-tiles * 64 N-groups = 16384 waves; 8 waves per WG
    bilstm_gemm_zx<<<16384 / 8, 256, 0, stream>>>(vfor, wxT, Zbuf);

    // Phase 2: persistent bidirectional scan, one grid barrier per step
    bilstm_scan<<<NWG_SCAN, 256, 0, stream>>>(w, Zbuf, hbuf, bar, out);
}
